// MultiHeadSelfAttention_55027120997040
// MI455X (gfx1250) — matrix operand order repaired and checked by reading, not compiled, not run on hardware
//
#include <hip/hip_runtime.h>
#include <hip/hip_bf16.h>

// ---------------------------------------------------------------------------
// MI455X (gfx1250, CDNA5) fused MHA+RoPE.
// bf16 WMMA (f32 accum) for all GEMMs + flash attention.
// TDM (tensor_load_to_lds) stages K tiles with hardware LDS padding.
// wave32; 256-thread blocks = 8 waves.
// ---------------------------------------------------------------------------

typedef __attribute__((ext_vector_type(16))) __bf16 v16bf;
typedef __attribute__((ext_vector_type(8)))  __bf16 v8bf;
typedef __attribute__((ext_vector_type(8)))  float  v8f;
typedef __attribute__((ext_vector_type(4)))  float  v4f;
typedef __attribute__((ext_vector_type(4)))  unsigned int v4u;
typedef __attribute__((ext_vector_type(8)))  int    v8i;
typedef __attribute__((ext_vector_type(4)))  int    v4i;

#define NUM_HEADS 16
#define HEAD_DIM  64
#define BATCH     4
#define SEQ       2048
#define EMB       1024
#define MTOT      (BATCH * SEQ)   // 8192 tokens

__device__ __forceinline__ v8f wmma_bf16(v16bf a, v16bf b, v8f c) {
  // D = A(16x32 bf16) * B(32x16 bf16) + C(16x16 f32)
  return __builtin_amdgcn_wmma_f32_16x16x32_bf16(
      false, a, false, b, (short)0, c, false, false);
}

union V16U { v16bf v; struct { v8bf lo; v8bf hi; } p; };
union BF2U { unsigned u; struct { __bf16 x; __bf16 y; } h; };

__device__ __forceinline__ v16bf ld16(const __bf16* plo, const __bf16* phi) {
  V16U u;
  u.p.lo = *(const v8bf*)plo;   // 16B -> ds_load_b128 / global_load_b128
  u.p.hi = *(const v8bf*)phi;
  return u.v;
}

__device__ __forceinline__ v8f zero8() {
  v8f z = {0.f, 0.f, 0.f, 0.f, 0.f, 0.f, 0.f, 0.f};
  return z;
}

// 16-lane butterfly step via ds_swizzle immediate (offset[14:10]=xor, [4:0]=and)
template<int XOR>
__device__ __forceinline__ float swz_xor(float x) {
  int i = __builtin_amdgcn_ds_swizzle(__float_as_int(x), (XOR << 10) | 0x1F);
  return __int_as_float(i);
}

// ---------------------------------------------------------------------------
// TDM: load a 64(keys) x 64(d) bf16 tile into LDS with 8-half row padding
// (pad 4 DWORDs after every 32 DWORDs -> LDS row stride 72 halves).
// D# packing per CDNA5 ISA ch.8 (group0 128b, group1 256b; 2D tensor).
// 6-arg builtin form: (g0, g1, g2, g3, g4, cpol).
// ---------------------------------------------------------------------------
__device__ __forceinline__ void tdm_load_tile64x64(const __bf16* gptr,
                                                   unsigned ldsOff) {
  unsigned long long ga = (unsigned long long)(uintptr_t)gptr;
  v4u g0;
  g0[0] = 1u;                                   // count=1 (valid descriptor)
  g0[1] = ldsOff;                               // lds_addr (bytes)
  g0[2] = (unsigned)(ga & 0xFFFFFFFFu);         // global_addr[31:0]
  g0[3] = (unsigned)((ga >> 32) & 0x01FFFFFFu)  // global_addr[56:32]
        | (2u << 30);                           // type=2 ("image")
  v8i g1;
  g1[0] = (int)((1u << 16)      // data_size = 1 -> 2 bytes
              | (1u << 20)      // pad_enable
              | (4u << 22)      // pad_interval: 32 DWORDs (= one 128B row)
              | (3u << 25));    // pad_amount: 4 DWORDs (= 8 halves)
  g1[1] = (int)((unsigned)HEAD_DIM << 16);   // tensor_dim0 = 64 (bits 79:48 lo)
  g1[2] = (int)((unsigned)SEQ << 16);        // tensor_dim1 = 2048 (bits 111:80 lo)
  g1[3] = (int)((unsigned)HEAD_DIM << 16);   // tile_dim0 = 64 (bits 127:112)
  g1[4] = 64;                                // tile_dim1 = 64 keys (bits 143:128)
  g1[5] = HEAD_DIM;                          // tensor_dim0_stride = 64 (bits 207:160 lo)
  g1[6] = 0;                                 // stride0 hi / tensor_dim1_stride lo
  g1[7] = 0;
  v4i z4 = {0, 0, 0, 0};                     // groups 2/3 unused (2D tensor)
  v8i z8 = {0, 0, 0, 0, 0, 0, 0, 0};
  __builtin_amdgcn_tensor_load_to_lds(g0, g1, z4, z4, z8, 0);
}

// ---------------------------------------------------------------------------
// GEMM: D[8192,1024] = op(A[8192,1024]) * W[1024,1024], epilogue bias*scale.
// Block tile 128x64, BK=64 (2 WMMA K-steps per stage). 8 waves @ 32x32.
// ---------------------------------------------------------------------------
template<bool A_IS_F32, bool OUT_IS_F32>
__global__ __launch_bounds__(256) void gemm_kernel(
    const void* __restrict__ Av, const float* __restrict__ W,
    const float* __restrict__ bias, const float* __restrict__ scale,
    void* __restrict__ Dv)
{
  constexpr int BM = 128, BN = 64, BK = 64;
  constexpr int LDA = BK + 8;   // 72 halves
  constexpr int LDB = BK + 8;
  __shared__ __bf16 As[BM * LDA];   // [row][k]
  __shared__ __bf16 Bt[BN * LDB];   // [n][k]  (transposed W tile)

  const int tid  = threadIdx.x;
  const int wave = tid >> 5, lane = tid & 31;
  const int mg = wave >> 1, ng = wave & 1;     // 4x2 wave grid
  const int lm = lane & 15, lhi = lane >> 4;
  const int khalf = lhi * 8;                   // A-frag K sub-offset
  const int koff  = lhi * 16;                  // unused (B-frags also use khalf)
  const int rowBase = blockIdx.x * BM;
  const int colBase = blockIdx.y * BN;

  v8f acc[2][2];
  acc[0][0] = zero8(); acc[0][1] = zero8();
  acc[1][0] = zero8(); acc[1][1] = zero8();

  const float*  Af = (const float*)Av;
  const __bf16* Ab = (const __bf16*)Av;

  for (int kt = 0; kt < EMB; kt += BK) {
    // ---- stage A tile (128x64 halves), 4 chunks of 8 per thread ----
#pragma unroll
    for (int c = 0; c < 4; c++) {
      int chunk = tid * 4 + c;          // 0..1023
      int r  = chunk >> 3;
      int kc = (chunk & 7) * 8;
      if (A_IS_F32) {
        const float* s = Af + (size_t)(rowBase + r) * EMB + kt + kc;
        v4f f0 = *(const v4f*)s;
        v4f f1 = *(const v4f*)(s + 4);
        __bf16* d = &As[r * LDA + kc];
#pragma unroll
        for (int i = 0; i < 4; i++) { d[i] = (__bf16)f0[i]; d[4 + i] = (__bf16)f1[i]; }
      } else {
        v8bf v = *(const v8bf*)(Ab + (size_t)(rowBase + r) * EMB + kt + kc);
        *(v8bf*)&As[r * LDA + kc] = v;
      }
    }
    // ---- stage B tile (64x64 f32) transposed -> Bt[n][k] bf16 ----
#pragma unroll
    for (int c = 0; c < 2; c++) {
      int chunk = tid * 2 + c;          // 0..511
      int k  = chunk >> 3;              // 0..63
      int nc = (chunk & 7) * 8;         // 0..56
      const float* s = W + (size_t)(kt + k) * EMB + colBase + nc;
      v4f f0 = *(const v4f*)s;
      v4f f1 = *(const v4f*)(s + 4);
#pragma unroll
      for (int j = 0; j < 4; j++) {
        Bt[(nc + j) * LDB + k]     = (__bf16)f0[j];
        Bt[(nc + 4 + j) * LDB + k] = (__bf16)f1[j];
      }
    }
    if (kt + BK < EMB)  // speculative prefetch of next W tile row
      __builtin_prefetch((const void*)(W + (size_t)(kt + BK + (tid >> 3)) * EMB + colBase), 0, 0);
    __syncthreads();

    // ---- 2 K-substeps x 4 WMMAs ----
#pragma unroll
    for (int ks = 0; ks < BK; ks += 32) {
      v16bf afr[2], bfr[2];
#pragma unroll
      for (int mi = 0; mi < 2; mi++) {
        const __bf16* p = &As[(mg * 32 + 16 * mi + lm) * LDA + ks];
        afr[mi] = ld16(p + khalf, p + khalf + 16);
      }
#pragma unroll
      for (int nj = 0; nj < 2; nj++) {
        const __bf16* p = &Bt[(ng * 32 + 16 * nj + lm) * LDB + ks];
        bfr[nj] = ld16(p + khalf, p + khalf + 16);
      }
#pragma unroll
      for (int mi = 0; mi < 2; mi++)
#pragma unroll
        for (int nj = 0; nj < 2; nj++)
          acc[mi][nj] = wmma_bf16(afr[mi], bfr[nj], acc[mi][nj]);
    }
    __syncthreads();
  }

  // ---- epilogue: (acc + bias) * scale ----
#pragma unroll
  for (int mi = 0; mi < 2; mi++) {
#pragma unroll
    for (int nj = 0; nj < 2; nj++) {
      int gn = colBase + ng * 32 + 16 * nj + lm;
      float bs = bias[gn], sc = scale[gn];
#pragma unroll
      for (int r = 0; r < 8; r++) {
        int gm = rowBase + mg * 32 + 16 * mi + r + 8 * lhi;
        float v = (acc[mi][nj][r] + bs) * sc;
        if (OUT_IS_F32) {
          ((float*)Dv)[(size_t)gm * EMB + gn] = v;
        } else {
          int b = gm >> 11, s = gm & (SEQ - 1);
          int h = gn >> 6,  d = gn & (HEAD_DIM - 1);
          ((__bf16*)Dv)[(((size_t)(b * NUM_HEADS + h)) * SEQ + s) * HEAD_DIM + d] = (__bf16)v;
        }
      }
    }
  }
}

// ---------------------------------------------------------------------------
// RoPE on Q and K (in place, packed 32-bit bf16-pair accesses).
// ---------------------------------------------------------------------------
__global__ __launch_bounds__(256) void rope_kernel(
    __bf16* __restrict__ Q, __bf16* __restrict__ Km, const int* __restrict__ pos)
{
  int t  = blockIdx.x * blockDim.x + threadIdx.x;
  int dp = t & 31;
  int s  = (t >> 5) & (SEQ - 1);
  int bh = t >> 16;                 // 0..63
  int b  = bh >> 4;

  float p    = (float)pos[b * SEQ + s];
  float freq = __powf(10000.f, -(float)(2 * dp) * (1.f / (float)HEAD_DIM));
  float sn, cs;
  __sincosf(p * freq, &sn, &cs);

  size_t idx = ((size_t)bh * SEQ + s) * HEAD_DIM + 2 * dp;

  BF2U q; q.u = *(const unsigned*)&Q[idx];
  float q0 = (float)q.h.x, q1 = (float)q.h.y;
  BF2U qo; qo.h.x = (__bf16)(q0 * cs - q1 * sn); qo.h.y = (__bf16)(q0 * sn + q1 * cs);
  *(unsigned*)&Q[idx] = qo.u;

  BF2U k; k.u = *(const unsigned*)&Km[idx];
  float k0 = (float)k.h.x, k1 = (float)k.h.y;
  BF2U ko; ko.h.x = (__bf16)(k0 * cs - k1 * sn); ko.h.y = (__bf16)(k0 * sn + k1 * cs);
  *(unsigned*)&Km[idx] = ko.u;
}

// ---------------------------------------------------------------------------
// Flash attention: grid (S/128, B*H). Each wave owns 16 q-rows; Q held in
// registers as two A-frags. 64-key tiles: K staged via TDM (hardware-padded
// rows), V staged transposed by VALU. Row-sums via P @ ones WMMA; row-max
// via ds_swizzle butterflies. 18 WMMAs per tile.
// ---------------------------------------------------------------------------
__global__ __launch_bounds__(256) void attn_kernel(
    const __bf16* __restrict__ Q, const __bf16* __restrict__ Km,
    const __bf16* __restrict__ V, __bf16* __restrict__ ctx)
{
  constexpr int KT  = 64;            // keys per tile
  constexpr int LDK = HEAD_DIM + 8;  // 72 (matches TDM pad: 4 DW per 32 DW)
  constexpr int LDV = KT + 8;        // 72
  constexpr int LDP = KT + 8;        // 72
  __shared__ __bf16 Kt[KT * LDK];         // [key][d]   (TDM destination)
  __shared__ __bf16 Vt[HEAD_DIM * LDV];   // [d][key]   (transposed)
  __shared__ __bf16 Ps[8 * 16 * LDP];     // per-wave P staging (C->A layout)

  const int tid  = threadIdx.x;
  const int wave = tid >> 5, lane = tid & 31;
  const int lm = lane & 15, lhi = lane >> 4;
  const int khalf = lhi * 8, koff = lhi * 16;
  const int bh   = blockIdx.y;                   // b*16 + h
  const size_t base = (size_t)bh * SEQ * HEAD_DIM;
  const int qrow0 = blockIdx.x * 128 + wave * 16;

  // Q fragments for this wave's 16 rows (held for the whole kernel)
  const __bf16* qp = Q + base + (size_t)(qrow0 + lm) * HEAD_DIM;
  v16bf qa0 = ld16(qp + khalf,      qp + khalf + 16);       // d 0..31
  v16bf qa1 = ld16(qp + 32 + khalf, qp + 32 + khalf + 16);  // d 32..63

  // all-ones B fragment for row-sum WMMA
  V16U uo;
#pragma unroll
  for (int i = 0; i < 8; i++) { uo.p.lo[i] = (__bf16)1.0f; uo.p.hi[i] = (__bf16)1.0f; }
  const v16bf onesB = uo.v;

  v8f o[4];
  o[0] = zero8(); o[1] = zero8(); o[2] = zero8(); o[3] = zero8();
  float mrow[8], lrow[8];
#pragma unroll
  for (int r = 0; r < 8; r++) { mrow[r] = -3.0e38f; lrow[r] = 0.f; }

  const unsigned ktOff = (unsigned)(uintptr_t)(void*)Kt;   // LDS byte offset

  for (int kt0 = 0; kt0 < SEQ; kt0 += KT) {
    // ---- K tile via Tensor Data Mover (one issue per block, wave 0) ----
    if (wave == 0) {
      tdm_load_tile64x64(Km + base + (size_t)kt0 * HEAD_DIM, ktOff);
    }
    // ---- V tile transposed by all 8 waves in parallel with the DMA ----
#pragma unroll
    for (int c = 0; c < 2; c++) {
      int chunk = tid * 2 + c;          // 0..511
      int key = chunk >> 3, dc = (chunk & 7) * 8;
      v8bf vv = *(const v8bf*)(V + base + (size_t)(kt0 + key) * HEAD_DIM + dc);
#pragma unroll
      for (int j = 0; j < 8; j++) Vt[(dc + j) * LDV + key] = vv[j];
    }
    if (wave == 0) __builtin_amdgcn_s_wait_tensorcnt(0);
    __syncthreads();

    // ---- scores: 4 frags x (2 WMMAs over d) ----
    v8f sf[4];
#pragma unroll
    for (int f = 0; f < 4; f++) {
      const __bf16* kp = &Kt[(16 * f + lm) * LDK];
      v16bf kb0 = ld16(kp + khalf,      kp + khalf + 16);       // d 0..31
      v16bf kb1 = ld16(kp + 32 + khalf, kp + 32 + khalf + 16);  // d 32..63
      v8f z = zero8();
      z = wmma_bf16(qa0, kb0, z);
      z = wmma_bf16(qa1, kb1, z);
      sf[f] = z;
    }

    // ---- online softmax: max via swizzle butterfly, exp, rescale O ----
#pragma unroll
    for (int r = 0; r < 8; r++) {
      float s0 = sf[0][r] * 0.125f;   // 1/sqrt(64)
      float s1 = sf[1][r] * 0.125f;
      float s2 = sf[2][r] * 0.125f;
      float s3 = sf[3][r] * 0.125f;
      float mx = fmaxf(fmaxf(s0, s1), fmaxf(s2, s3));
      mx = fmaxf(mx, swz_xor<8>(mx));
      mx = fmaxf(mx, swz_xor<4>(mx));
      mx = fmaxf(mx, swz_xor<2>(mx));
      mx = fmaxf(mx, swz_xor<1>(mx));
      float nm    = fmaxf(mrow[r], mx);
      float alpha = __expf(mrow[r] - nm);
      mrow[r] = nm;
      sf[0][r] = __expf(s0 - nm);
      sf[1][r] = __expf(s1 - nm);
      sf[2][r] = __expf(s2 - nm);
      sf[3][r] = __expf(s3 - nm);
      lrow[r] *= alpha;
      o[0][r] *= alpha; o[1][r] *= alpha; o[2][r] *= alpha; o[3][r] *= alpha;
    }

    // ---- C-frag -> A-frag via per-wave LDS round trip ----
    __bf16* pb = &Ps[wave * 16 * LDP];
#pragma unroll
    for (int r = 0; r < 8; r++) {
      int row = r + 8 * lhi;
      pb[row * LDP + lm]      = (__bf16)sf[0][r];
      pb[row * LDP + 16 + lm] = (__bf16)sf[1][r];
      pb[row * LDP + 32 + lm] = (__bf16)sf[2][r];
      pb[row * LDP + 48 + lm] = (__bf16)sf[3][r];
    }
    const __bf16* pr = pb + lm * LDP;
    v16bf pa0 = ld16(pr + khalf,      pr + khalf + 16);       // keys 0..31
    v16bf pa1 = ld16(pr + 32 + khalf, pr + 32 + khalf + 16);  // keys 32..63

    // ---- row sums via P @ ones (replaces shuffle-reduce) ----
    v8f rs = zero8();
    rs = wmma_bf16(pa0, onesB, rs);
    rs = wmma_bf16(pa1, onesB, rs);
#pragma unroll
    for (int r = 0; r < 8; r++) lrow[r] += rs[r];

    // ---- O += P(16x64) x V(64x64): 8 WMMAs ----
#pragma unroll
    for (int j = 0; j < 4; j++) {
      const __bf16* vp = &Vt[(16 * j + lm) * LDV];
      v16bf vb0 = ld16(vp + khalf,      vp + khalf + 16);        // keys 0..31
      v16bf vb1 = ld16(vp + 32 + khalf, vp + 32 + khalf + 16);   // keys 32..63
      o[j] = wmma_bf16(pa0, vb0, o[j]);
      o[j] = wmma_bf16(pa1, vb1, o[j]);
    }
    __syncthreads();
  }

  // ---- normalize and store ctx (b, s, h*64+d) bf16 ----
  const int b = bh >> 4, h = bh & 15;
#pragma unroll
  for (int r = 0; r < 8; r++) {
    float inv = 1.0f / lrow[r];
    int row = qrow0 + r + 8 * lhi;
    size_t rb = ((size_t)(b * SEQ + row)) * EMB + h * HEAD_DIM;
#pragma unroll
    for (int j = 0; j < 4; j++)
      ctx[rb + 16 * j + lm] = (__bf16)(o[j][r] * inv);
  }
}

// ---------------------------------------------------------------------------
// Launcher. Inputs: x, positions, Wq,bq, Wk,bk, Wv,bv, Wo,bo, q_s,k_s,v_s,o_s
// ws layout: Qh | Kh | Vh (B,H,S,D bf16) | ctx (B,S,E bf16) = 4 x 16 MB.
// ---------------------------------------------------------------------------
extern "C" void kernel_launch(void* const* d_in, const int* in_sizes, int n_in,
                              void* d_out, int out_size, void* d_ws, size_t ws_size,
                              hipStream_t stream) {
  (void)in_sizes; (void)n_in; (void)out_size; (void)ws_size;
  const float* x   = (const float*)d_in[0];
  const int*   pos = (const int*)d_in[1];
  const float* Wq  = (const float*)d_in[2];
  const float* bq  = (const float*)d_in[3];
  const float* Wk  = (const float*)d_in[4];
  const float* bk  = (const float*)d_in[5];
  const float* Wv  = (const float*)d_in[6];
  const float* bv  = (const float*)d_in[7];
  const float* Wo  = (const float*)d_in[8];
  const float* bo  = (const float*)d_in[9];
  const float* qs  = (const float*)d_in[10];
  const float* ks  = (const float*)d_in[11];
  const float* vs  = (const float*)d_in[12];
  const float* os  = (const float*)d_in[13];

  const size_t HBUF = (size_t)MTOT * EMB;   // 8,388,608 bf16 elements each
  __bf16* Qh  = (__bf16*)d_ws;
  __bf16* Kh  = Qh + HBUF;
  __bf16* Vh  = Kh + HBUF;
  __bf16* ctx = Vh + HBUF;

  dim3 gGemm(MTOT / 128, EMB / 64), blk(256);
  // Q/K/V projections: f32 x, f32 W -> bf16 (B,H,S,D), bias*scale fused
  gemm_kernel<true, false><<<gGemm, blk, 0, stream>>>(x, Wq, bq, qs, Qh);
  gemm_kernel<true, false><<<gGemm, blk, 0, stream>>>(x, Wk, bk, ks, Kh);
  gemm_kernel<true, false><<<gGemm, blk, 0, stream>>>(x, Wv, bv, vs, Vh);

  // RoPE on Q and K
  rope_kernel<<<(BATCH * NUM_HEADS * SEQ * 32) / 256, blk, 0, stream>>>(Qh, Kh, pos);

  // flash attention -> ctx (B,S,E) bf16
  attn_kernel<<<dim3(SEQ / 128, BATCH * NUM_HEADS), blk, 0, stream>>>(Qh, Kh, Vh, ctx);

  // output projection: bf16 ctx, f32 Wo -> f32 out, bias*scale fused
  gemm_kernel<false, true><<<gGemm, blk, 0, stream>>>(ctx, Wo, bo, os, d_out);
}
